// Net_72919954752303
// MI455X (gfx1250) — compile-verified, matrix operand order checked by reading, not run on hardware
//
#include <hip/hip_runtime.h>
#include <hip/hip_bf16.h>

// Net: SchNet-like GNN, fused per-block message passing with bf16 WMMA.
// B=2, N=512, D=F=128, G=32, T=3.

#define BATCH  2
#define NATOM  512
#define FDIM   128
#define NG     32
#define NT     3
#define BN     (BATCH*NATOM)
#define NTILE  (BATCH*(NATOM/16))   // 64 i-tiles
#define JSPLIT 8                    // j-range split factor (occupancy)
#define JLEN   (NATOM/JSPLIT)       // 64 j's per workgroup

#if __has_builtin(__builtin_amdgcn_global_load_async_to_lds_b128) && \
    __has_builtin(__builtin_amdgcn_s_wait_asynccnt)
#define USE_ASYNC_LDS 1
#else
#define USE_ASYNC_LDS 0
#endif

typedef __attribute__((ext_vector_type(16))) __bf16 bf16x16;
typedef __attribute__((ext_vector_type(8)))  __bf16 bf16x8;
typedef __attribute__((ext_vector_type(8)))  float  f32x8;
typedef int v4i __attribute__((vector_size(4 * sizeof(int))));
typedef __attribute__((address_space(1))) v4i* gv4ip;   // global int4*
typedef __attribute__((address_space(3))) v4i* lv4ip;   // LDS int4*

// shifted softplus via raw v_exp_f32 / v_log_f32 (log2 domain):
// ssp(x) = max(x,0) + ln2 * (log2(1 + 2^(-|x|*log2e)) - 1)
__device__ __forceinline__ float ssp(float x) {
  const float LOG2E = 1.4426950408889634f;
  const float LN2   = 0.6931471805599453f;
  float t = __builtin_amdgcn_exp2f(-fabsf(x) * LOG2E);
  float l = __builtin_amdgcn_logf(1.0f + t);
  return fmaxf(x, 0.0f) + LN2 * (l - 1.0f);
}

__device__ __forceinline__ void wave_lds_sync() {
  // intra-wave LDS RAW handoff (cross-lane): drain DScnt (CDNA5 split counters)
  asm volatile("s_wait_dscnt 0" ::: "memory");
}

__device__ __forceinline__ bf16x16 combine8(bf16x8 lo, bf16x8 hi) {
  union { bf16x16 v; bf16x8 h[2]; } u;
  u.h[0] = lo; u.h[1] = hi;
  return u.v;
}

// ---------------------------------------------------------------------------
// Weight prep: bf16 + transpose so WMMA B-fragments are two contiguous 16B
// chunks per lane.  W1T[t][f][g] = W1s[t][g][f];  W2T[t][fo][fi] = W2s[t][fi][fo]
// ---------------------------------------------------------------------------
__global__ __launch_bounds__(256) void prep_weights(
    const float* __restrict__ W1s, const float* __restrict__ W2s,
    __bf16* __restrict__ W1T, __bf16* __restrict__ W2T) {
  const int t = blockIdx.x;
  for (int o = threadIdx.x; o < FDIM * NG; o += 256) {
    int f = o / NG, g = o % NG;
    W1T[t * FDIM * NG + o] = (__bf16)W1s[t * NG * FDIM + g * FDIM + f];
  }
  for (int o = threadIdx.x; o < FDIM * FDIM; o += 256) {
    int fo = o >> 7, fi = o & 127;
    W2T[t * FDIM * FDIM + o] = (__bf16)W2s[t * FDIM * FDIM + fi * FDIM + fo];
  }
}

// h[bn][d] = embed[r[bn]][d]
__global__ __launch_bounds__(128) void embed_kernel(
    const int* __restrict__ r, const float* __restrict__ embed,
    float* __restrict__ h) {
  const int bn = blockIdx.x;
  h[(size_t)bn * FDIM + threadIdx.x] =
      embed[(size_t)r[bn] * FDIM + threadIdx.x];
}

// rf[bn][f] = sum_d h[bn][d] * Waf[d][f]
__global__ __launch_bounds__(128) void rf_kernel(
    const float* __restrict__ h, const float* __restrict__ Waf,
    float* __restrict__ rf) {
  __shared__ float hrow[FDIM];
  const int bn = blockIdx.x, f = threadIdx.x;
  hrow[f] = h[(size_t)bn * FDIM + f];
  __syncthreads();
  float s = 0.0f;
  #pragma unroll 8
  for (int d = 0; d < FDIM; ++d) s += hrow[d] * Waf[d * FDIM + f];
  rf[(size_t)bn * FDIM + f] = s;
}

// ---------------------------------------------------------------------------
// Fused filter + message partial sum:
//   grid = NTILE * JSPLIT.  workgroup: one (b, 16-atom i-tile, j-chunk of 64).
//   8 waves, wave owns 2 i-rows.  Per 16-j tile: dist/mask/gaussians ->
//   WMMA1 (K=32 bf16) -> ssp -> WMMA2 (K=128 bf16) -> +b2, *mask, *rf[j],
//   reduce over j.  rf tiles are double-buffered via async global->LDS DMA
//   (ASYNCcnt pipelined against the WMMAs).  Partial y -> ypart[jchunk][tile].
// ---------------------------------------------------------------------------
__global__ __launch_bounds__(256, 1) void message_kernel(
    const float* __restrict__ xyz, const float* __restrict__ rfg,
    const __bf16* __restrict__ W1T, const float* __restrict__ b1,
    const __bf16* __restrict__ W2T, const float* __restrict__ b2,
    float* __restrict__ ypart) {
  __shared__ __align__(16) float  e_lds[16 * 16];
  __shared__ __align__(16) float  m_lds[16 * 16];
  __shared__ __align__(16) float  rfbuf[2][16 * FDIM];      // double-buffered
  __shared__ __align__(16) __bf16 stage_lds[8][16 * FDIM];  // per-wave ssp(GEMM1)

  const int blk    = blockIdx.x;
  const int tile   = blk / JSPLIT;       // 0..63
  const int jchunk = blk % JSPLIT;
  const int b      = tile >> 5;          // 32 i-tiles per batch
  const int i0     = (tile & 31) << 4;
  const int tid    = threadIdx.x;
  const int wave   = tid >> 5;
  const int lane   = tid & 31;
  const int half   = lane >> 4;
  const int ln     = lane & 15;

  const float GW  = 5.0f / 31.0f;                              // gaussian width
  const float GC2 = (-0.5f / (GW * GW)) * 1.4426950408889634f; // coeff * log2e

  // rf tile fetch: 16 rows x 512B = 512 16-byte chunks; 2 per thread.
  auto issue_rf = [&](int jrow0, int buf) {
    #pragma unroll
    for (int k = 0; k < 2; ++k) {
      const int c   = tid + k * 256;
      const int row = c >> 5;            // 32 chunks per 128-float row
      const int col = (c & 31) * 4;      // float offset
      const float* src = rfg + ((size_t)b * NATOM + jrow0 + row) * FDIM + col;
      float* dst = &rfbuf[buf][row * FDIM + col];
#if USE_ASYNC_LDS
      __builtin_amdgcn_global_load_async_to_lds_b128((gv4ip)src, (lv4ip)dst,
                                                     0, 0);
#else
      *(float4*)dst = *(const float4*)src;
#endif
    }
  };

  // Stage-1 B fragments (W1^T, 32x128) resident in VGPRs for the whole kernel.
  // 16-bit B layout: lane = column n; VGPR0-3: K = 8*half+0..7,
  // VGPR4-7: K = 16+8*half+0..7.
  bf16x16 w1b[8];
  #pragma unroll
  for (int nt = 0; nt < 8; ++nt) {
    const __bf16* row = W1T + (nt * 16 + ln) * NG;
    bf16x8 lo = *(const bf16x8*)(row + 8 * half);
    bf16x8 hi = *(const bf16x8*)(row + 16 + 8 * half);
    w1b[nt] = combine8(lo, hi);
  }

  float yacc[2][8];
  #pragma unroll
  for (int a = 0; a < 2; ++a)
    #pragma unroll
    for (int n2 = 0; n2 < 8; ++n2) yacc[a][n2] = 0.0f;

  const int it0 = wave * 2;
  const int jbase = jchunk * JLEN;

  issue_rf(jbase, 0);                    // prologue: tile 0 in flight

  for (int jt = 0; jt < JLEN; jt += 16) {
    const int j0  = jbase + jt;
    const int cur = (jt >> 4) & 1;
    __syncthreads();   // everyone done with prev e/m and rfbuf[cur^1]
    {  // pairwise distances + cutoff mask (one pair per thread)
      const int ii = tid >> 4, jj = tid & 15;
      const float* pi = xyz + ((size_t)b * NATOM + i0 + ii) * 3;
      const float* pj = xyz + ((size_t)b * NATOM + j0 + jj) * 3;
      float dx = pi[0] - pj[0], dy = pi[1] - pj[1], dz = pi[2] - pj[2];
      float dsq = dx * dx + dy * dy + dz * dz;
      float msk = (dsq <= 100.0f && dsq != 0.0f) ? 1.0f : 0.0f;
      e_lds[tid] = sqrtf(dsq * msk + 1e-8f);
      m_lds[tid] = msk;
    }
    // kick DMA for the next tile, then drain this tile's DMA (in-order per
    // wave: <=2 outstanding leaves only the just-issued pair in flight)
    const bool more = (jt + 16 < JLEN);
    if (more) issue_rf(j0 + 16, cur ^ 1);
#if USE_ASYNC_LDS
    if (more) __builtin_amdgcn_s_wait_asynccnt(2);
    else      __builtin_amdgcn_s_wait_asynccnt(0);
#endif
    __syncthreads();   // e/m + rfbuf[cur] now valid workgroup-wide
    const float* rf_lds = rfbuf[cur];

    #pragma unroll
    for (int iti = 0; iti < 2; ++iti) {
      const int it = it0 + iti;   // i-row this wave handles
      // Stage 1 A fragment: M = j_local = ln (both halves); 16-bit A layout:
      // elems 0..7 -> K=8*half+0..7, elems 8..15 -> K=16+8*half+0..7
      const float ev = e_lds[it * 16 + ln];
      bf16x16 afrag;
      #pragma unroll
      for (int ee = 0; ee < 16; ++ee) {
        const int k = (ee < 8) ? (8 * half + ee) : (16 + 8 * half + (ee - 8));
        const float d = ev - (float)k * GW;
        afrag[ee] = (__bf16)__builtin_amdgcn_exp2f(GC2 * d * d);
      }
      __bf16* stg = &stage_lds[wave][0];
      #pragma unroll
      for (int nt = 0; nt < 8; ++nt) {
        f32x8 c = {0.f, 0.f, 0.f, 0.f, 0.f, 0.f, 0.f, 0.f};
        c = __builtin_amdgcn_wmma_f32_16x16x32_bf16(
            false, afrag, false, w1b[nt], (short)0, c, false, false);
        const float bias = b1[nt * 16 + ln];
        #pragma unroll
        for (int v = 0; v < 8; ++v) {  // C layout: M = v + 8*half, N = ln
          const int m = v + 8 * half;
          stg[m * FDIM + nt * 16 + ln] = (__bf16)ssp(c[v] + bias);
        }
      }
      wave_lds_sync();

      // Stage 2: (16 x 128) @ (128 x 128), K chained over 4 WMMAs per n-tile
      #pragma unroll
      for (int nt = 0; nt < 8; ++nt) {
        f32x8 c2 = {0.f, 0.f, 0.f, 0.f, 0.f, 0.f, 0.f, 0.f};
        #pragma unroll
        for (int kc = 0; kc < 4; ++kc) {
          const int k0 = kc * 32;
          bf16x8 alo = *(const bf16x8*)(stg + ln * FDIM + k0 + 8 * half);
          bf16x8 ahi = *(const bf16x8*)(stg + ln * FDIM + k0 + 16 + 8 * half);
          const __bf16* wrow = W2T + (nt * 16 + ln) * FDIM + k0;
          bf16x8 blo = *(const bf16x8*)(wrow + 8 * half);
          bf16x8 bhi = *(const bf16x8*)(wrow + 16 + 8 * half);
          c2 = __builtin_amdgcn_wmma_f32_16x16x32_bf16(
              false, combine8(alo, ahi), false, combine8(blo, bhi),
              (short)0, c2, false, false);
        }
        const float bias2 = b2[nt * 16 + ln];
        float acc = yacc[iti][nt];
        #pragma unroll
        for (int v = 0; v < 8; ++v) {
          const int m = v + 8 * half;  // j_local
          const float wv = (c2[v] + bias2) * m_lds[it * 16 + m];
          acc += wv * rf_lds[m * FDIM + nt * 16 + ln];
        }
        yacc[iti][nt] = acc;
      }
    }
  }

  // fold lane-half partials (m 0..7 vs 8..15) and write partial y tile
  float* yout = ypart + ((size_t)jchunk * NTILE + tile) * (16 * FDIM);
  #pragma unroll
  for (int iti = 0; iti < 2; ++iti) {
    #pragma unroll
    for (int nt = 0; nt < 8; ++nt) {
      float v = yacc[iti][nt];
      v += __shfl_xor(v, 16, 32);
      if (half == 0) yout[(it0 + iti) * FDIM + nt * 16 + ln] = v;
    }
  }
}

// ---------------------------------------------------------------------------
// Combine JSPLIT partial y tiles, Dense1+ssp, Dense2, residual.
// grid = NTILE, 256 threads.
// ---------------------------------------------------------------------------
__global__ __launch_bounds__(256) void dense_kernel(
    const float* __restrict__ ypart,
    const float* __restrict__ h_in, float* __restrict__ h_out,
    const float* __restrict__ Wd1, const float* __restrict__ bd1,
    const float* __restrict__ Wd2, const float* __restrict__ bd2) {
  __shared__ float y_lds[16 * FDIM];
  __shared__ float d1_lds[16 * FDIM];
  const int tile = blockIdx.x;           // 0..63
  const int b    = tile >> 5;
  const int i0   = (tile & 31) << 4;
  const int tid  = threadIdx.x;

  for (int o = tid; o < 16 * FDIM; o += 256) {
    float s = 0.0f;
    #pragma unroll
    for (int jc = 0; jc < JSPLIT; ++jc)
      s += ypart[((size_t)jc * NTILE + tile) * (16 * FDIM) + o];
    y_lds[o] = s;
  }
  __syncthreads();

  for (int o = tid; o < 16 * FDIM; o += 256) {
    const int ii = o >> 7, f = o & 127;
    float s = bd1[f];
    const float* yr = &y_lds[ii * FDIM];
    #pragma unroll 8
    for (int k = 0; k < FDIM; ++k) s += yr[k] * Wd1[k * FDIM + f];
    d1_lds[o] = ssp(s);
  }
  __syncthreads();
  for (int o = tid; o < 16 * FDIM; o += 256) {
    const int ii = o >> 7, f = o & 127;
    float s = bd2[f];
    const float* dr = &d1_lds[ii * FDIM];
    #pragma unroll 8
    for (int k = 0; k < FDIM; ++k) s += dr[k] * Wd2[k * FDIM + f];
    const size_t gi = ((size_t)b * NATOM + i0 + ii) * FDIM + f;
    h_out[gi] = h_in[gi] + s;
  }
}

// out[b] = sum_n ( ssp(h[n] @ aw1 + b1) @ aw2 + b2 )
__global__ __launch_bounds__(256) void readout_kernel(
    const float* __restrict__ h, const float* __restrict__ aw1w,
    const float* __restrict__ aw1b, const float* __restrict__ aw2w,
    const float* __restrict__ aw2b, float* __restrict__ out) {
  __shared__ float red[256];
  const int b = blockIdx.x, tid = threadIdx.x;
  float acc = 0.0f;
  for (int n = tid; n < NATOM; n += 256) {
    const float* hr = &h[((size_t)b * NATOM + n) * FDIM];
    float val = aw2b[0];
    #pragma unroll 4
    for (int o = 0; o < FDIM / 2; ++o) {
      float s = aw1b[o];
      for (int d = 0; d < FDIM; ++d) s += hr[d] * aw1w[d * (FDIM / 2) + o];
      val += ssp(s) * aw2w[o];
    }
    acc += val;
  }
  red[tid] = acc;
  __syncthreads();
  for (int s = 128; s > 0; s >>= 1) {
    if (tid < s) red[tid] += red[tid + s];
    __syncthreads();
  }
  if (tid == 0) out[b] = red[0];
}

extern "C" void kernel_launch(void* const* d_in, const int* in_sizes, int n_in,
                              void* d_out, int out_size, void* d_ws,
                              size_t ws_size, hipStream_t stream) {
  (void)in_sizes; (void)n_in; (void)out_size; (void)ws_size;
  const int*   r     = (const int*)d_in[0];
  const float* xyz   = (const float*)d_in[1];
  const float* embed = (const float*)d_in[2];
  const float* W1s   = (const float*)d_in[3];
  const float* b1s   = (const float*)d_in[4];
  const float* W2s   = (const float*)d_in[5];
  const float* b2s   = (const float*)d_in[6];
  const float* Wafs  = (const float*)d_in[7];
  const float* Wd1s  = (const float*)d_in[8];
  const float* bd1s  = (const float*)d_in[9];
  const float* Wd2s  = (const float*)d_in[10];
  const float* bd2s  = (const float*)d_in[11];
  const float* aw1w  = (const float*)d_in[12];
  const float* aw1b  = (const float*)d_in[13];
  const float* aw2w  = (const float*)d_in[14];
  const float* aw2b  = (const float*)d_in[15];
  float* out = (float*)d_out;

  char* ws = (char*)d_ws;
  float*  h_a   = (float*)ws;  ws += (size_t)BN * FDIM * sizeof(float);
  float*  h_b   = (float*)ws;  ws += (size_t)BN * FDIM * sizeof(float);
  float*  rf    = (float*)ws;  ws += (size_t)BN * FDIM * sizeof(float);
  float*  ypart = (float*)ws;  ws += (size_t)JSPLIT * NTILE * 16 * FDIM * sizeof(float);
  __bf16* W1T   = (__bf16*)ws; ws += (size_t)NT * FDIM * NG * sizeof(__bf16);
  __bf16* W2T   = (__bf16*)ws; ws += (size_t)NT * FDIM * FDIM * sizeof(__bf16);

  prep_weights<<<NT, 256, 0, stream>>>(W1s, W2s, W1T, W2T);
  embed_kernel<<<BN, 128, 0, stream>>>(r, embed, h_a);

  float* hin = h_a;
  float* hout = h_b;
  for (int t = 0; t < NT; ++t) {
    rf_kernel<<<BN, 128, 0, stream>>>(hin, Wafs + (size_t)t * FDIM * FDIM, rf);
    message_kernel<<<NTILE * JSPLIT, 256, 0, stream>>>(
        xyz, rf,
        W1T + (size_t)t * FDIM * NG,   b1s + (size_t)t * FDIM,
        W2T + (size_t)t * FDIM * FDIM, b2s + (size_t)t * FDIM,
        ypart);
    dense_kernel<<<NTILE, 256, 0, stream>>>(
        ypart, hin, hout,
        Wd1s + (size_t)t * FDIM * FDIM, bd1s + (size_t)t * FDIM,
        Wd2s + (size_t)t * FDIM * FDIM, bd2s + (size_t)t * FDIM);
    float* tmp = hin; hin = hout; hout = tmp;
  }
  readout_kernel<<<BATCH, 256, 0, stream>>>(hin, aw1w, aw1b, aw2w, aw2b, out);
}
